// SparseBlock_33638183862315
// MI455X (gfx1250) — matrix operand order repaired and checked by reading, not compiled, not run on hardware
//
#include <hip/hip_runtime.h>
#include <hip/hip_bf16.h>

#define CIN   64
#define COUT  64
#define KTAPS 9
#define EPSB  1e-4f
#define LEAKF 0.333f

typedef __attribute__((ext_vector_type(16))) __bf16 v16bf;
typedef __attribute__((ext_vector_type(8)))  float  v8f;

// Round-to-nearest-even float -> bf16 bit pattern (storage as u16 so host code
// never touches __bf16).
__device__ inline unsigned short f2bf(float x) {
    union { float f; unsigned int u; } c; c.f = x;
    unsigned int u = c.u;
    unsigned int r = u + 0x7FFFu + ((u >> 16) & 1u);
    return (unsigned short)(r >> 16);
}

// ---------------------------------------------------------------------------
// Kernel 1: convert W[k][cin][cout] (f32) into the CDNA5 WMMA B-fragment
// layout for v_wmma_f32_16x16x32_bf16 (B is 32x16 bf16, 8 VGPRs/lane):
//   lane L holds column (L&15)+16*nb, elem e=2j+h -> K (L>>4)*8 + (e&7) + 16*(e>>3) (+32*s).
// Flat fragment index: f = ((k*2+s)*4+nb)*512 + lane*16 + j*2 + h
// ---------------------------------------------------------------------------
__global__ void prep_wfrag(const float* __restrict__ W,
                           unsigned short* __restrict__ wfrag) {
    int t = blockIdx.x * blockDim.x + threadIdx.x;
    const int total = KTAPS * 2 * 4 * 32 * 16; // 36864
    for (int f = t; f < total; f += blockDim.x * gridDim.x) {
        int h    = f & 1;
        int j    = (f >> 1) & 7;
        int lane = (f >> 4) & 31;
        int nb   = (f >> 9) & 3;
        int s    = (f >> 11) & 1;
        int k    = f >> 12;
        int cin  = (lane >> 4) * 8 + s * 32 + ((2 * j + h) & 7) + ((2 * j + h) >> 3) * 16;
        int cout = (lane & 15) + 16 * nb;
        wfrag[f] = f2bf(W[(k * CIN + cin) * COUT + cout]);
    }
}

// ---------------------------------------------------------------------------
// Kernel 2: gather -> bf16 WMMA -> accumulate, one 16x64 tile per wave.
// A-fragment (16-bit A 16x32): lane L owns row L&15; channel base
// kb=(L>>4)*8; elems 0..7 = chans kb..kb+7, elems 8..15 = chans kb+16..kb+23
// (+32*s per k-step). Matches CDNA5 ISA VGPR tables.
// ---------------------------------------------------------------------------
__global__ __launch_bounds__(256) void sconv_wmma(
    const float* __restrict__ feats,
    const unsigned short* __restrict__ wfrag,
    const int* __restrict__ nbr,
    const unsigned char* __restrict__ valid,
    float* __restrict__ y,            // d_out used as y (N x 64 f32)
    float* __restrict__ partials,     // [nblocks][128]: 64 sums, 64 sumsq
    int n)
{
    __shared__ float s_sum[8 * 64];
    __shared__ float s_sq [8 * 64];

    const int tid  = threadIdx.x;
    const int wave = tid >> 5;
    const int lane = tid & 31;
    const int r    = lane & 15;
    const int hi   = lane >> 4;
    const int kb   = hi * 8;

    const int m0 = (blockIdx.x * 8 + wave) * 16;
    const int m  = m0 + r;
    const int row_ok = (m < n) ? 1 : 0;
    const int mc = (m < n) ? m : 0;

    // Rulebook preload: addresses are always safe (mc clamped), so load
    // unconditionally and combine with bitwise AND -> no EXEC divergence,
    // all 18 loads issue back-to-back.
    int idxs[KTAPS];
    int okv [KTAPS];
#pragma unroll
    for (int k = 0; k < KTAPS; ++k) {
        idxs[k] = nbr[(size_t)mc * KTAPS + k];
        int vb  = (int)valid[(size_t)mc * KTAPS + k];
        okv[k]  = row_ok & (vb != 0 ? 1 : 0);
    }

    v8f acc0 = {}, acc1 = {}, acc2 = {}, acc3 = {};

    for (int k = 0; k < KTAPS; ++k) {
        const float* src = feats + (size_t)idxs[k] * CIN;
        if (k + 1 < KTAPS)   // CDNA5 global_prefetch_b8 of next tap's row
            __builtin_prefetch(feats + (size_t)idxs[k + 1] * CIN, 0, 1);
        const bool ok = (okv[k] != 0);

#pragma unroll
        for (int s = 0; s < 2; ++s) {
            const int base = s * 32 + kb;
            float4 c0 = *(const float4*)(src + base);
            float4 c1 = *(const float4*)(src + base + 4);
            float4 c2 = *(const float4*)(src + base + 16);
            float4 c3 = *(const float4*)(src + base + 20);
            float fv[16] = { c0.x,c0.y,c0.z,c0.w, c1.x,c1.y,c1.z,c1.w,
                             c2.x,c2.y,c2.z,c2.w, c3.x,c3.y,c3.z,c3.w };
            v16bf a;
#pragma unroll
            for (int e = 0; e < 16; ++e)
                a[e] = (__bf16)fv[e];
            a = ok ? a : v16bf{};     // select on 8 packed dwords, no branch

            const unsigned short* wp =
                wfrag + ((size_t)((k * 2 + s) * 4) * 32 + lane) * 16;
            v16bf b0 = *(const v16bf*)(wp + 0 * 512);
            v16bf b1 = *(const v16bf*)(wp + 1 * 512);
            v16bf b2 = *(const v16bf*)(wp + 2 * 512);
            v16bf b3 = *(const v16bf*)(wp + 3 * 512);

            acc0 = __builtin_amdgcn_wmma_f32_16x16x32_bf16(false, a, false, b0, (short)0, acc0, false, false);
            acc1 = __builtin_amdgcn_wmma_f32_16x16x32_bf16(false, a, false, b1, (short)0, acc1, false, false);
            acc2 = __builtin_amdgcn_wmma_f32_16x16x32_bf16(false, a, false, b2, (short)0, acc2, false, false);
            acc3 = __builtin_amdgcn_wmma_f32_16x16x32_bf16(false, a, false, b3, (short)0, acc3, false, false);
        }
    }

    // Store C tile (lane L: VGPR v -> row hi*8+v, col (L&15)+16*nb) and build
    // deterministic per-block BN partial sums (shfl + LDS, no atomics).
    v8f accs[4] = { acc0, acc1, acc2, acc3 };
#pragma unroll
    for (int j = 0; j < 4; ++j) {
        float lsum = 0.0f, lsq = 0.0f;
#pragma unroll
        for (int v = 0; v < 8; ++v) {
            float val = accs[j][v];
            int row = m0 + hi * 8 + v;
            if (row < n) y[(size_t)row * COUT + (r + 16 * j)] = val;
            lsum += val;
            lsq  += val * val;
        }
        lsum += __shfl_xor(lsum, 16, 32);
        lsq  += __shfl_xor(lsq,  16, 32);
        if (hi == 0) {
            s_sum[wave * 64 + 16 * j + r] = lsum;
            s_sq [wave * 64 + 16 * j + r] = lsq;
        }
    }
    __syncthreads();
    if (tid < 128) {
        int c = tid & 63;
        const float* arr = (tid < 64) ? s_sum : s_sq;
        float accv = 0.0f;
#pragma unroll
        for (int w = 0; w < 8; ++w) accv += arr[w * 64 + c];
        partials[(size_t)blockIdx.x * 128 + tid] = accv;
    }
}

// ---------------------------------------------------------------------------
// Kernel 3: deterministic reduction of per-block partials -> stats[128]
// ---------------------------------------------------------------------------
__global__ void reduce_stats(const float* __restrict__ partials,
                             float* __restrict__ stats, int nblocks) {
    int t = threadIdx.x;  // 128 threads, 1 block
    float acc = 0.0f;
    for (int b = 0; b < nblocks; ++b)
        acc += partials[(size_t)b * 128 + t];
    stats[t] = acc;
}

// ---------------------------------------------------------------------------
// Kernel 4: BatchNorm + LeakyReLU, in place on y (= d_out), float4 vectorized
// ---------------------------------------------------------------------------
__global__ __launch_bounds__(256) void bn_leaky(
    float* __restrict__ y, const float* __restrict__ stats,
    const float* __restrict__ gamma, const float* __restrict__ beta, int n)
{
    __shared__ float s_scale[COUT];
    __shared__ float s_bias [COUT];
    if (threadIdx.x < COUT) {
        int c = threadIdx.x;
        float invn = 1.0f / (float)n;
        float mean = stats[c] * invn;
        float var  = fmaxf(stats[64 + c] * invn - mean * mean, 0.0f);
        float sc   = gamma[c] * rsqrtf(var + EPSB);
        s_scale[c] = sc;
        s_bias [c] = beta[c] - mean * sc;
    }
    __syncthreads();

    size_t total4 = (size_t)n * (COUT / 4);
    for (size_t q = (size_t)blockIdx.x * blockDim.x + threadIdx.x; q < total4;
         q += (size_t)gridDim.x * blockDim.x) {
        int cb = ((int)(q & 15)) * 4;
        float4 v = ((float4*)y)[q];
        float o;
        o = v.x * s_scale[cb + 0] + s_bias[cb + 0]; v.x = (o >= 0.f) ? o : LEAKF * o;
        o = v.y * s_scale[cb + 1] + s_bias[cb + 1]; v.y = (o >= 0.f) ? o : LEAKF * o;
        o = v.z * s_scale[cb + 2] + s_bias[cb + 2]; v.z = (o >= 0.f) ? o : LEAKF * o;
        o = v.w * s_scale[cb + 3] + s_bias[cb + 3]; v.w = (o >= 0.f) ? o : LEAKF * o;
        ((float4*)y)[q] = v;
    }
}

extern "C" void kernel_launch(void* const* d_in, const int* in_sizes, int n_in,
                              void* d_out, int out_size, void* d_ws, size_t ws_size,
                              hipStream_t stream) {
    const float* feats          = (const float*)d_in[0];
    const float* W              = (const float*)d_in[1];
    const float* gamma          = (const float*)d_in[2];
    const float* beta           = (const float*)d_in[3];
    const int*   nbr            = (const int*)d_in[4];
    const unsigned char* valid  = (const unsigned char*)d_in[5];
    float* y = (float*)d_out;

    int n = in_sizes[0] / CIN;

    // ws layout: [0, 73728) bf16 W fragments; [73728, 74240) stats;
    //            [74240, ...) per-block partials (nblocks * 512 B)
    unsigned short* wfrag = (unsigned short*)d_ws;
    float* stats    = (float*)((char*)d_ws + 73728);
    float* partials = (float*)((char*)d_ws + 73728 + 512);

    int nblocks = (n + 127) / 128;

    hipLaunchKernelGGL(prep_wfrag, dim3(144), dim3(256), 0, stream, W, wfrag);
    hipLaunchKernelGGL(sconv_wmma, dim3(nblocks), dim3(256), 0, stream,
                       feats, wfrag, nbr, valid, y, partials, n);
    hipLaunchKernelGGL(reduce_stats, dim3(1), dim3(128), 0, stream,
                       partials, stats, nblocks);

    long total4 = (long)n * (COUT / 4);
    long want   = (total4 + 255) / 256;
    int bnblocks = (int)((want < 16384) ? want : 16384);
    if (bnblocks < 1) bnblocks = 1;
    hipLaunchKernelGGL(bn_leaky, dim3(bnblocks), dim3(256), 0, stream,
                       y, stats, gamma, beta, n);
}
